// Disease_Guide_ROI_24550033064603
// MI455X (gfx1250) — matrix operand order repaired and checked by reading, not compiled, not run on hardware
//
#include <hip/hip_runtime.h>
#include <hip/hip_bf16.h>

// ---------------------------------------------------------------------------
// N==1 makes attention degenerate (softmax over 1 element == 1), so:
//   v      = kv_w[90:180] @ x + kv_b[90:180]            (per row, 90x90)
//   weight = w0
//   3x:  x1 = v*weight; gi = W_ih@x1+b_ih; gh = W_hh@weight+b_hh; GRU gates
//   out    = proj_w @ x1 + proj_b
// Batched over 131072 rows -> WMMA GEMMs.
// M = 32 batch rows per wave (2 m-tiles): every B fragment feeds TWO v_wmma.
// r/z gates accumulate both their GEMMs into ONE WMMA accumulator chain.
// Biases folded into K-slot 90 (staging col 90 == 1.0, weight row 90 == bias)
// so all WMMA chains start from C = inline 0.
// vfrag kept as packed f16 (frees 48 VGPRs) so the 6 B fragments per (t,kc)
// can stay live simultaneously -> pipelined loads instead of wait-to-zero
// stalls seen in round 4's GRU body.
// __launch_bounds__(128, 2): cap at 512 VGPRs -> 2 waves/SIMD.
// ---------------------------------------------------------------------------

#define DIMC   90
#define COLP   96      // padded channel dim (K and N for 90-wide mats)
#define ROW3   288     // padded 3C gate dim (3 * 96)
#define WAVES  4       // waves per block
#define MT     2       // 16-row m-tiles per wave -> 32 rows/wave, 128/block

typedef _Float16 v16h __attribute__((ext_vector_type(16)));
typedef _Float16 v8h  __attribute__((ext_vector_type(8)));
typedef _Float16 v2h  __attribute__((ext_vector_type(2)));
typedef float    v8f  __attribute__((ext_vector_type(8)));

// ws layout (f16 elements):
//   Wt_kv [96*96]  : k<90: kv_w[(90+n)*90+k] ; k==90: kv_b[90+n]   ; else 0
//   Wt_ih [96*288] : k<90: w_ih[(g*90+c)*90+k]; k==90: b_ih[g*90+c]; else 0
//   Wt_hh [96*288] : same with w_hh / b_hh
//   Wt_pj [96*96]  : k<90: proj_w[n*90+k]   ; k==90: proj_b[n]     ; else 0
#define NKV  (96*96)
#define NIH  (96*288)
#define KBIAS 90

__device__ __forceinline__ float sigmoidf_(float v) {
    return 1.0f / (1.0f + __expf(-v));
}
__device__ __forceinline__ float tanhf_(float v) {
    float e = __expf(-2.0f * v);
    return (1.0f - e) / (1.0f + e);
}

__device__ __forceinline__ v2h pack_f16(float a, float b) {
    return __builtin_bit_cast(v2h, __builtin_amdgcn_cvt_pkrtz(a, b));
}

__device__ __forceinline__ v8f wmma_f16(v16h a, v16h b, v8f c) {
    // D = A(16x32 f16) x B(32x16 f16) + C(16x16 f32)
    return __builtin_amdgcn_wmma_f32_16x16x32_f16(
        /*neg_a=*/false, a, /*neg_b=*/false, b,
        /*c_mod=*/(short)0, c, /*reuse_a=*/false, /*reuse_b=*/false);
}

// A fragment (16x32, f16): lane = {m = L%16, half = L/16};
// vector element e<8  -> K = 32*kc + 8*half + e
// vector element e>=8 -> K = 32*kc + 16 + 8*half + (e-8)
__device__ __forceinline__ v16h load_a_lds(const _Float16* s, int kc, int lane) {
    int m = lane & 15, half = lane >> 4;
    const _Float16* p = s + m * COLP + kc * 32 + half * 8;
    v8h lo = *(const v8h*)p;
    v8h hi = *(const v8h*)(p + 16);
    v16h a;
#pragma unroll
    for (int i = 0; i < 8; ++i) { a[i] = lo[i]; a[i + 8] = hi[i]; }
    return a;
}

// B fragment (32x16, f16): lane -> N = 16*nt + L%16, half = L/16;
// element e<8 -> K = 32*kc + 8*half + e, e>=8 -> K = 32*kc + 16 + 8*half + (e-8)
// (Wt stored as [k][n], row length rowlen)
__device__ __forceinline__ v16h load_b_glb(const _Float16* __restrict__ W,
                                           int rowlen, int kc, int nt, int lane) {
    const _Float16* p = W + (size_t)(kc * 32 + (lane >> 4) * 8) * rowlen + nt * 16 + (lane & 15);
    v16h b;
#pragma unroll
    for (int i = 0; i < 8; ++i) { b[i] = p[(size_t)i * rowlen]; b[i + 8] = p[(size_t)(16 + i) * rowlen]; }
    return b;
}

// ---------------------------------------------------------------------------
// Prep: transpose/pad weights to f16 B-fragment layout; bias in row k==90.
// ---------------------------------------------------------------------------
__global__ void prep_weights(const float* __restrict__ kv_w, const float* __restrict__ kv_b,
                             const float* __restrict__ w_ih, const float* __restrict__ w_hh,
                             const float* __restrict__ b_ih, const float* __restrict__ b_hh,
                             const float* __restrict__ proj_w, const float* __restrict__ proj_b,
                             _Float16* __restrict__ ws) {
    _Float16* Wkv = ws;
    _Float16* Wih = Wkv + NKV;
    _Float16* Whh = Wih + NIH;
    _Float16* Wpj = Whh + NIH;

    int idx = blockIdx.x * blockDim.x + threadIdx.x;
    if (idx < NKV) {
        int k = idx / 96, n = idx % 96;
        float v = 0.0f;
        if (n < DIMC) {
            if (k < DIMC)        v = kv_w[(size_t)(DIMC + n) * DIMC + k];
            else if (k == KBIAS) v = kv_b[DIMC + n];
        }
        Wkv[idx] = (_Float16)v;
    } else if (idx < NKV + NIH) {
        int j = idx - NKV;
        int k = j / ROW3, n = j % ROW3, g = n / 96, c = n % 96;
        float v = 0.0f;
        if (c < DIMC) {
            if (k < DIMC)        v = w_ih[(size_t)(g * DIMC + c) * DIMC + k];
            else if (k == KBIAS) v = b_ih[g * DIMC + c];
        }
        Wih[j] = (_Float16)v;
    } else if (idx < NKV + 2 * NIH) {
        int j = idx - NKV - NIH;
        int k = j / ROW3, n = j % ROW3, g = n / 96, c = n % 96;
        float v = 0.0f;
        if (c < DIMC) {
            if (k < DIMC)        v = w_hh[(size_t)(g * DIMC + c) * DIMC + k];
            else if (k == KBIAS) v = b_hh[g * DIMC + c];
        }
        Whh[j] = (_Float16)v;
    } else if (idx < 2 * NKV + 2 * NIH) {
        int j = idx - NKV - 2 * NIH;
        int k = j / 96, n = j % 96;
        float v = 0.0f;
        if (n < DIMC) {
            if (k < DIMC)        v = proj_w[(size_t)n * DIMC + k];
            else if (k == KBIAS) v = proj_b[n];
        }
        Wpj[j] = (_Float16)v;
    }
}

// ---------------------------------------------------------------------------
// Main fused kernel: 32 batch rows per wave (2 m-tiles sharing B fragments).
// C/D fragment layout: lane L -> n = L%16; VGPR j -> m = j + 8*(L/16).
// ---------------------------------------------------------------------------
__global__ __launch_bounds__(WAVES * 32, 2) void
fused_gru_attn(const float* __restrict__ x, const float* __restrict__ w0,
               const _Float16* __restrict__ ws, float* __restrict__ out) {
    const _Float16* Wkv = ws;
    const _Float16* Wih = Wkv + NKV;
    const _Float16* Whh = Wih + NIH;
    const _Float16* Wpj = Whh + NIH;

    __shared__ __align__(16) _Float16 s_x1[WAVES][MT * 16 * COLP];
    __shared__ __align__(16) _Float16 s_w [WAVES][MT * 16 * COLP];

    const int lane = threadIdx.x & 31;
    const int wave = threadIdx.x >> 5;
    const int nlo  = lane & 15;
    const int mhi  = (lane >> 4) * 8;
    const size_t m0 = ((size_t)blockIdx.x * WAVES + wave) * (MT * 16);

    _Float16* sx = s_x1[wave];
    _Float16* sw = s_w[wave];

    // stage x rows (f32 global -> f16 LDS); col 90 = 1.0 (bias slot), 91..95 = 0
    for (int idx = lane; idx < MT * 16 * COLP; idx += 32) {
        int m = idx / COLP, c = idx % COLP;
        float v = (c < DIMC) ? x[(m0 + m) * DIMC + c] : (c == KBIAS ? 1.0f : 0.0f);
        sx[idx] = (_Float16)v;
    }
    __syncthreads();

    // Stage A: v = x @ Wt_kv (+kv_b via K-slot 90); C starts at inline 0.
    // v kept as packed f16 pairs (element (p,h) -> VGPR j = 2p+h).
    v2h vfrag[MT][6][4];
    {
        v16h A[MT][3];
#pragma unroll
        for (int u = 0; u < MT; ++u)
#pragma unroll
            for (int kc = 0; kc < 3; ++kc)
                A[u][kc] = load_a_lds(sx + u * 16 * COLP, kc, lane);
#pragma unroll
        for (int t = 0; t < 6; ++t) {
            v8f acc[MT] = {};
#pragma unroll
            for (int kc = 0; kc < 3; ++kc) {
                v16h Bf = load_b_glb(Wkv, COLP, kc, t, lane);
#pragma unroll
                for (int u = 0; u < MT; ++u) acc[u] = wmma_f16(A[u][kc], Bf, acc[u]);
            }
#pragma unroll
            for (int u = 0; u < MT; ++u)
#pragma unroll
                for (int p = 0; p < 4; ++p)
                    vfrag[u][t][p] = pack_f16(acc[u][2 * p], acc[u][2 * p + 1]);
        }
    }

    // weight init = w0 (broadcast over batch rows)
    float wfrag[MT][6][8];
#pragma unroll
    for (int t = 0; t < 6; ++t) {
        int c = t * 16 + nlo;
        float v = (c < DIMC) ? w0[c] : 0.0f;
#pragma unroll
        for (int u = 0; u < MT; ++u)
#pragma unroll
            for (int j = 0; j < 8; ++j) wfrag[u][t][j] = v;
    }

    v16h Ax[MT][3];   // x1 A-fragments (kept for the final projection)
#pragma unroll 1
    for (int it = 0; it < 3; ++it) {
        // x1 = v * weight (f16); stage x1 and weight (fragment -> row-major)
        // staging column 90 is forced to 1.0 (bias slot)
#pragma unroll
        for (int u = 0; u < MT; ++u) {
            _Float16* sxu = sx + u * 16 * COLP;
            _Float16* swu = sw + u * 16 * COLP;
#pragma unroll
            for (int t = 0; t < 6; ++t) {
                int cbase = t * 16 + nlo;
                bool isb = (cbase == KBIAS);   // only possible in tile t==5
#pragma unroll
                for (int j = 0; j < 8; ++j) {
                    int m = j + mhi;
                    _Float16 wv = isb ? (_Float16)1.0f : (_Float16)wfrag[u][t][j];
                    _Float16 xv = isb ? (_Float16)1.0f
                                      : (_Float16)(vfrag[u][t][j >> 1][j & 1] * wv);
                    sxu[m * COLP + cbase] = xv;
                    swu[m * COLP + cbase] = wv;
                }
            }
        }
        __syncthreads();

        v16h Aw[MT][3];
#pragma unroll
        for (int u = 0; u < MT; ++u)
#pragma unroll
            for (int kc = 0; kc < 3; ++kc) {
                Ax[u][kc] = load_a_lds(sx + u * 16 * COLP, kc, lane);
                Aw[u][kc] = load_a_lds(sw + u * 16 * COLP, kc, lane);
            }

        // GRU gates. 3C padded to 288 so gates c, c+C, c+2C share lane mapping.
        // r/z: single accumulator carries (x1@Wih + w@Whh); biases via K-slot.
        // All 6 B fragments of a (t,kc) step stay live -> pipelined loads.
#pragma unroll
        for (int t = 0; t < 6; ++t) {
            v8f ar[MT] = {}, az[MT] = {}, ain[MT] = {}, ahn[MT] = {};
#pragma unroll
            for (int kc = 0; kc < 3; ++kc) {
                v16h B0 = load_b_glb(Wih, ROW3, kc, t,      lane);
                v16h B1 = load_b_glb(Whh, ROW3, kc, t,      lane);
                v16h B2 = load_b_glb(Wih, ROW3, kc, t + 6,  lane);
                v16h B3 = load_b_glb(Whh, ROW3, kc, t + 6,  lane);
                v16h B4 = load_b_glb(Wih, ROW3, kc, t + 12, lane);
                v16h B5 = load_b_glb(Whh, ROW3, kc, t + 12, lane);
#pragma unroll
                for (int u = 0; u < MT; ++u) ar[u]  = wmma_f16(Ax[u][kc], B0, ar[u]);
#pragma unroll
                for (int u = 0; u < MT; ++u) ar[u]  = wmma_f16(Aw[u][kc], B1, ar[u]);
#pragma unroll
                for (int u = 0; u < MT; ++u) az[u]  = wmma_f16(Ax[u][kc], B2, az[u]);
#pragma unroll
                for (int u = 0; u < MT; ++u) az[u]  = wmma_f16(Aw[u][kc], B3, az[u]);
#pragma unroll
                for (int u = 0; u < MT; ++u) ain[u] = wmma_f16(Ax[u][kc], B4, ain[u]);
#pragma unroll
                for (int u = 0; u < MT; ++u) ahn[u] = wmma_f16(Aw[u][kc], B5, ahn[u]);
            }
#pragma unroll
            for (int u = 0; u < MT; ++u)
#pragma unroll
                for (int j = 0; j < 8; ++j) {
                    float r = sigmoidf_(ar[u][j]);
                    float z = sigmoidf_(az[u][j]);
                    float n = tanhf_(ain[u][j] + r * ahn[u][j]);
                    wfrag[u][t][j] = (1.0f - z) * n + z * wfrag[u][t][j];
                }
        }
        __syncthreads();
    }

    // out = x1 @ Wt_pj (+proj_b via K-slot); Ax still holds last x1
#pragma unroll
    for (int t = 0; t < 6; ++t) {
        v8f acc[MT] = {};
#pragma unroll
        for (int kc = 0; kc < 3; ++kc) {
            v16h Bf = load_b_glb(Wpj, COLP, kc, t, lane);
#pragma unroll
            for (int u = 0; u < MT; ++u) acc[u] = wmma_f16(Ax[u][kc], Bf, acc[u]);
        }
        int c = t * 16 + nlo;
        if (c < DIMC) {
#pragma unroll
            for (int u = 0; u < MT; ++u)
#pragma unroll
                for (int j = 0; j < 8; ++j)
                    out[(m0 + u * 16 + j + mhi) * DIMC + c] = acc[u][j];
        }
    }
}

extern "C" void kernel_launch(void* const* d_in, const int* in_sizes, int n_in,
                              void* d_out, int out_size, void* d_ws, size_t ws_size,
                              hipStream_t stream) {
    const float* x      = (const float*)d_in[0];
    // d_in[1] = cls_out (dead: softmax over 1 element)
    const float* w0     = (const float*)d_in[2];
    const float* kv_w   = (const float*)d_in[3];
    const float* kv_b   = (const float*)d_in[4];
    // d_in[5]/d_in[6] = cls_w / cls_b (dead)
    const float* w_ih   = (const float*)d_in[7];
    const float* w_hh   = (const float*)d_in[8];
    const float* b_ih   = (const float*)d_in[9];
    const float* b_hh   = (const float*)d_in[10];
    const float* proj_w = (const float*)d_in[11];
    const float* proj_b = (const float*)d_in[12];
    float* out = (float*)d_out;

    const int Btot = in_sizes[0] / DIMC;   // 131072

    _Float16* ws = (_Float16*)d_ws;        // ~144 KB used
    const int prep_elems = 2 * NKV + 2 * NIH;
    prep_weights<<<(prep_elems + 255) / 256, 256, 0, stream>>>(
        kv_w, kv_b, w_ih, w_hh, b_ih, b_hh, proj_w, proj_b, ws);

    const int rows_per_block = WAVES * MT * 16;  // 128; Btot is a multiple of 128
    fused_gru_attn<<<Btot / rows_per_block, WAVES * 32, 0, stream>>>(x, w0, ws, out);
}
